// ScaledDotProductAttention_22737556865169
// MI455X (gfx1250) — compile-verified, matrix operand order checked by reading, not run on hardware
//
#include <hip/hip_runtime.h>
#include <math.h>

typedef __attribute__((ext_vector_type(16))) _Float16 v16h;
typedef __attribute__((ext_vector_type(8)))  _Float16 v8h;
typedef __attribute__((ext_vector_type(4)))  _Float16 v4h;
typedef __attribute__((ext_vector_type(8)))  float    v8f;
typedef __attribute__((ext_vector_type(4)))  float    v4f;

#define SEQ  2048
#define DIM  64
#define TQ   128          // queries per workgroup
#define WQ   16           // queries per wave
#define TK   64           // keys per inner block
#define NW   8            // waves per workgroup
#define NBLK (SEQ / TK)   // 32

// log2(e)/sqrt(64) folded score scale; mask bias in log2 domain
#define SC2    0.18033688f
#define BIAS2 -1.4426950409e30f

union Op16 { v16h v; v8h h[2]; };

__global__ __launch_bounds__(256, 1)
void fa_wmma_kernel(const float* __restrict__ Q, const float* __restrict__ K,
                    const float* __restrict__ V, const int* __restrict__ mask,
                    float* __restrict__ Out, int H)
{
    // Manual LDS layout (56 KB total):
    //   [    0, 16384) : Ks  double-buffered K tiles, f16 [2][TK*DIM], row-major (key, d)
    //   [16384, 32768) : Vt  double-buffered V tiles, f16 [2][DIM*TK], TRANSPOSED (d, key)
    //   [32768, 40960) : Bs  mask bias table, f32 [SEQ] (log2 domain)
    //   [40960, 57344) : Ps  per-wave P scratch f16 [NW][WQ*TK]; Q tile overlays this
    //                    region pre-loop (wave-local both ways, same-wave DS order).
    __shared__ alignas(16) unsigned char smem[57344];
    _Float16* Ks = (_Float16*)(smem);
    _Float16* Vt = (_Float16*)(smem + 16384);
    float*    Bs = (float*)   (smem + 32768);
    _Float16* Ps = (_Float16*)(smem + 40960);
    _Float16* Qs = (_Float16*)(smem + 40960);   // overlay, pre-loop only

    const int tid  = threadIdx.x;
    const int wave = tid >> 5;
    const int lane = tid & 31;
    const int nrow = lane & 15;              // N (or M) index this lane owns
    const int ko   = (lane < 16) ? 0 : 8;    // K-offset of this half-wave
    const int mbase = (lane >> 4) * 8;       // D-tile row base for this half-wave

    const int qtiles = SEQ / TQ;
    const int bh     = blockIdx.x / qtiles;
    const int qtile  = blockIdx.x % qtiles;
    const int b      = bh / H;

    const float* Qp = Q + (size_t)bh * SEQ * DIM + (size_t)qtile * TQ * DIM;
    const float* Kp = K + (size_t)bh * SEQ * DIM;
    const float* Vp = V + (size_t)bh * SEQ * DIM;
    const int*   Mp = mask + (size_t)b * SEQ;

    // ---- stage Q tile (f32->f16). Thread tid covers elements [tid*32, tid*32+32):
    // rows tid/2 -> wave w stages exactly its own 16 rows (wave-local). ----
    {
        const float* src = Qp + tid * 32;
        _Float16*    dst = Qs + tid * 32;
#pragma unroll
        for (int j = 0; j < 8; ++j) {
            v4f q = *(const v4f*)(src + j * 4);
            v4h h; h[0] = (_Float16)q[0]; h[1] = (_Float16)q[1];
                   h[2] = (_Float16)q[2]; h[3] = (_Float16)q[3];
            *(v4h*)(dst + j * 4) = h;
        }
    }

    // ---- Q A-operands: two 16x32 chunks, each = two contiguous 16B LDS loads ----
    Op16 qa[2];
    {
        const _Float16* qrow = &Qs[(wave * WQ + nrow) * DIM];
#pragma unroll
        for (int c = 0; c < 2; ++c) {
            qa[c].h[0] = *(const v8h*)(qrow + c * 32 + ko);
            qa[c].h[1] = *(const v8h*)(qrow + c * 32 + ko + 16);
        }
    }

    // ---- stage mask bias table (log2 domain), coalesced ----
    for (int i = tid; i < SEQ; i += 256)
        Bs[i] = Mp[i] ? 0.0f : BIAS2;

    // ---- prefetch registers for K/V staging ----
    // K: thread takes 16 contiguous floats (4 x float4), fully coalesced.
    // V: thread takes d = tid&63 fixed, 16 consecutive keys -> coalesced across
    //    lanes per j, and the transposed LDS store is contiguous (2 x b128).
    const int vd = tid & 63;
    const int vk0 = (tid >> 6) * 16;
    v4f  kreg[4];
    float vreg[16];

    auto prefetch = [&](int kb) {
        const float* kp = Kp + (size_t)kb * DIM + tid * 16;
#pragma unroll
        for (int j = 0; j < 4; ++j) kreg[j] = *(const v4f*)(kp + j * 4);
        const float* vp = Vp + (size_t)kb * DIM + (size_t)vk0 * DIM + vd;
#pragma unroll
        for (int j = 0; j < 16; ++j) vreg[j] = vp[j * DIM];
    };

    auto stage = [&](int buf) {
        _Float16* kd = Ks + buf * (TK * DIM) + tid * 16;
#pragma unroll
        for (int p = 0; p < 2; ++p) {
            v8h h;
#pragma unroll
            for (int j = 0; j < 8; ++j) h[j] = (_Float16)kreg[p * 2 + (j >> 2)][j & 3];
            *(v8h*)(kd + p * 8) = h;
        }
        _Float16* vdst = Vt + buf * (DIM * TK) + vd * TK + vk0;
#pragma unroll
        for (int p = 0; p < 2; ++p) {
            v8h h;
#pragma unroll
            for (int j = 0; j < 8; ++j) h[j] = (_Float16)vreg[p * 8 + j];
            *(v8h*)(vdst + p * 8) = h;
        }
    };

    const v8f vzero = {};
    v8f acc[4];
#pragma unroll
    for (int t = 0; t < 4; ++t) acc[t] = vzero;
    float mrow[8], lrow[8];
#pragma unroll
    for (int r = 0; r < 8; ++r) { mrow[r] = -INFINITY; lrow[r] = 0.0f; }

    // prologue: block 0 into buffer 0
    prefetch(0);
    stage(0);

#pragma unroll 1
    for (int i = 0; i < NBLK; ++i) {
        const int buf = i & 1;
        const int kb  = i * TK;
        __syncthreads();                 // buffer `buf` fully staged; prev reads of buf^1 done
        if (i + 1 < NBLK) prefetch(kb + TK);   // global loads fly under the compute below

        const _Float16* Kb = Ks + buf * (TK * DIM);
        const _Float16* Vb = Vt + buf * (DIM * TK);

        // ---- S = Q K^T : four 16(q) x 16(k) tiles, inner d = 64 ----
        v8f s[4];
#pragma unroll
        for (int t = 0; t < 4; ++t) {
            s[t] = vzero;
            const _Float16* krow = Kb + (t * 16 + nrow) * DIM;
#pragma unroll
            for (int c = 0; c < 2; ++c) {
                Op16 kop;
                kop.h[0] = *(const v8h*)(krow + c * 32 + ko);
                kop.h[1] = *(const v8h*)(krow + c * 32 + ko + 16);
                s[t] = __builtin_amdgcn_wmma_f32_16x16x32_f16(
                    false, qa[c].v, false, kop.v, (short)0, s[t], false, false);
            }
            const float bias = Bs[kb + t * 16 + nrow];
#pragma unroll
            for (int r = 0; r < 8; ++r) s[t][r] = fmaf(s[t][r], SC2, bias);
        }

        // ---- online softmax in log2 domain (rows spread over 16 lanes) ----
        float rescale[8];
#pragma unroll
        for (int r = 0; r < 8; ++r) {
            float mx = fmaxf(fmaxf(s[0][r], s[1][r]), fmaxf(s[2][r], s[3][r]));
#pragma unroll
            for (int off = 8; off >= 1; off >>= 1)
                mx = fmaxf(mx, __shfl_xor(mx, off, 32));
            const float mnew = fmaxf(mrow[r], mx);
            rescale[r] = __builtin_amdgcn_exp2f(mrow[r] - mnew);
            float sm = 0.0f;
#pragma unroll
            for (int t = 0; t < 4; ++t) {
                s[t][r] = __builtin_amdgcn_exp2f(s[t][r] - mnew);
                sm += s[t][r];
            }
#pragma unroll
            for (int off = 8; off >= 1; off >>= 1)
                sm += __shfl_xor(sm, off, 32);
            lrow[r] = lrow[r] * rescale[r] + sm;
            mrow[r] = mnew;
        }

        // ---- P: D-layout -> A-layout via per-wave LDS bounce ----
        _Float16* P = Ps + wave * (WQ * TK);
#pragma unroll
        for (int t = 0; t < 4; ++t)
#pragma unroll
            for (int r = 0; r < 8; ++r)
                P[(mbase + r) * TK + t * 16 + nrow] = (_Float16)s[t][r];
        Op16 pa[2];
        {
            const _Float16* prow = P + nrow * TK;
#pragma unroll
            for (int c = 0; c < 2; ++c) {
                pa[c].h[0] = *(const v8h*)(prow + c * 32 + ko);
                pa[c].h[1] = *(const v8h*)(prow + c * 32 + ko + 16);
            }
        }

        // ---- O = O*rescale + P V : four 16(q) x 16(d) tiles, inner 64 keys ----
#pragma unroll
        for (int nt = 0; nt < 4; ++nt) {
#pragma unroll
            for (int r = 0; r < 8; ++r) acc[nt][r] *= rescale[r];
            const _Float16* vrow = Vb + (nt * 16 + nrow) * TK;  // Vt row = d
#pragma unroll
            for (int c = 0; c < 2; ++c) {
                Op16 vb;
                vb.h[0] = *(const v8h*)(vrow + c * 32 + ko);
                vb.h[1] = *(const v8h*)(vrow + c * 32 + ko + 16);
                acc[nt] = __builtin_amdgcn_wmma_f32_16x16x32_f16(
                    false, pa[c].v, false, vb.v, (short)0, acc[nt], false, false);
            }
        }

        if (i + 1 < NBLK) stage(buf ^ 1);   // safe: all waves passed this iter's barrier
    }

    // ---- normalize and write out (coalesced 16-lane rows) ----
    float* Op = Out + (size_t)bh * SEQ * DIM + (size_t)(qtile * TQ + wave * WQ) * DIM;
#pragma unroll
    for (int nt = 0; nt < 4; ++nt)
#pragma unroll
        for (int r = 0; r < 8; ++r)
            Op[(mbase + r) * DIM + nt * 16 + nrow] = acc[nt][r] / lrow[r];
}

extern "C" void kernel_launch(void* const* d_in, const int* in_sizes, int n_in,
                              void* d_out, int out_size, void* d_ws, size_t ws_size,
                              hipStream_t stream) {
    const float* Q    = (const float*)d_in[0];
    const float* K    = (const float*)d_in[1];
    const float* V    = (const float*)d_in[2];
    const int*   mask = (const int*)d_in[3];
    float*       out  = (float*)d_out;

    const int BH = in_sizes[0] / (SEQ * DIM);   // B*H = 32
    const int B  = in_sizes[3] / SEQ;           // 4
    const int H  = BH / B;                      // 8

    dim3 grid(BH * (SEQ / TQ));                 // 512 workgroups
    fa_wmma_kernel<<<grid, 256, 0, stream>>>(Q, K, V, mask, out, H);
}
